// Transformer_60842506715598
// MI455X (gfx1250) — compile-verified, matrix operand order checked by reading, not run on hardware
//
#include <hip/hip_runtime.h>
#include <hip/hip_bf16.h>
#include <stdint.h>

// ---------------------------------------------------------------------------
// MoE transformer forward for MI455X (gfx1250, wave32, WMMA + TDM).
// All matmuls run on v_wmma_f32_16x16x32_f16 (f32 accumulate).
// Weight tiles are streamed into LDS by the Tensor Data Mover
// (tensor_load_to_lds, TENSORcnt) when the builtin is available.
// Expert FFN computed sparsely (top-2 routing) via grouped WMMA GEMMs.
// ---------------------------------------------------------------------------

typedef __attribute__((ext_vector_type(16))) _Float16     vhalf16;
typedef __attribute__((ext_vector_type(8)))  float        vfloat8;
typedef __attribute__((ext_vector_type(4)))  unsigned int v4u;
typedef __attribute__((ext_vector_type(8)))  int          v8i;
typedef __attribute__((ext_vector_type(4)))  int          v4i;

#define SEQ   1024
#define DIM   1024
#define NHEAD 16
#define NKV   4
#define HD    64
#define FFD   3584
#define NEXP  8
#define VOCAB 32000
#define WIN   512
#define NLAYR 2

#if defined(__has_builtin)
#  if __has_builtin(__builtin_amdgcn_tensor_load_to_lds)
#    define USE_TDM 1
#  else
#    define USE_TDM 0
#  endif
#else
#  define USE_TDM 0
#endif

__device__ __forceinline__ vfloat8 wmma_f16(vhalf16 a, vhalf16 b, vfloat8 c) {
    return __builtin_amdgcn_wmma_f32_16x16x32_f16(false, a, false, b,
                                                  (short)0, c, false, false);
}

// ---------------------------------------------------------------------------
// TDM: DMA one 32x128 fp32 tile (row stride = ldb elements) from global into
// LDS at byte offset lds_off, padded +1 DWORD per 128-DWORD row -> LDS row
// stride of 129 floats (bank-conflict-free column reads).
// D# layout per CDNA5 ISA ch.8 (group0: count/lds/global/type, group1: dims).
// This toolchain exposes the 6-arg builtin:
//   (v4u g0, v8i g1, v4i g2, v4i g3, v8i g4, i32 cpol)
// ---------------------------------------------------------------------------
#if USE_TDM
__device__ __forceinline__ void tdm_load_f32_tile(const float* gptr,
                                                  unsigned lds_off, int ldb) {
    unsigned long long ga = (unsigned long long)(__SIZE_TYPE__)gptr;
    v4u g0;
    g0[0] = 1u;                                   // count=1 (valid), user mode
    g0[1] = lds_off;                              // lds_addr (bytes)
    g0[2] = (unsigned)ga;                         // global_addr[31:0]
    g0[3] = (unsigned)((ga >> 32) & 0x1ffffffu)   // global_addr[56:32]
            | (2u << 30);                         // type = 2 ("image")
    v8i g1;
    g1[0] = (int)((2u << 16)                      // data_size = 4B
                  | (1u << 20)                    // pad_enable
                  | (6u << 22));                  // pad_interval = 128 DWORDs
                                                  // pad_amount   = 1 DWORD
    g1[1] = (int)(128u << 16);                    // tensor_dim0 lo = 128
    g1[2] = (int)(32u << 16);                     // dim0 hi=0 | tensor_dim1 lo = 32
    g1[3] = (int)(128u << 16);                    // dim1 hi=0 | tile_dim0 = 128
    g1[4] = (int)32u;                             // tile_dim1 = 32, tile_dim2 = 0
    g1[5] = (int)ldb;                             // tensor_dim0_stride lo
    g1[6] = 0;
    g1[7] = 0;
    v4i z4 = {};
    v8i z8 = {};
    __builtin_amdgcn_tensor_load_to_lds(g0, g1, z4, z4, z8, 0);
}
#endif

// Cooperative fallback: 256 threads stage a 32x128 f32 tile into LDS (stride 129)
__device__ __forceinline__ void stage_b_coop(const float* __restrict__ B, int ldb,
                                             int t, float* __restrict__ Bsf) {
    int brow = t >> 3, bc = (t & 7) * 16;
    const float4* p = (const float4*)(B + (size_t)brow * ldb + bc);
    #pragma unroll
    for (int i = 0; i < 4; i++) {
        float4 v = p[i];
        Bsf[brow * 129 + bc + 4 * i + 0] = v.x;
        Bsf[brow * 129 + bc + 4 * i + 1] = v.y;
        Bsf[brow * 129 + bc + 4 * i + 2] = v.z;
        Bsf[brow * 129 + bc + 4 * i + 3] = v.w;
    }
}

// ===========================================================================
// Embedding gather
// ===========================================================================
__global__ void k_embed(const int* __restrict__ tokens,
                        const float* __restrict__ emb,
                        float* __restrict__ x) {
    int s = blockIdx.x;
    int tok = tokens[s];
    const float* src = emb + (size_t)tok * DIM;
    float* dst = x + (size_t)s * DIM;
    for (int d = threadIdx.x; d < DIM; d += blockDim.x) dst[d] = src[d];
}

// ===========================================================================
// RMSNorm
// ===========================================================================
__global__ void k_rmsnorm(const float* __restrict__ x,
                          const float* __restrict__ w,
                          float* __restrict__ y) {
    int s = blockIdx.x, t = threadIdx.x;
    __shared__ float red[256];
    const float* xr = x + (size_t)s * DIM;
    float ss = 0.f;
    for (int d = t; d < DIM; d += 256) { float v = xr[d]; ss += v * v; }
    red[t] = ss;
    __syncthreads();
    for (int o = 128; o > 0; o >>= 1) {
        if (t < o) red[t] += red[t + o];
        __syncthreads();
    }
    float r = rsqrtf(red[0] / (float)DIM + 1e-5f);
    float* yr = y + (size_t)s * DIM;
    for (int d = t; d < DIM; d += 256) yr[d] = xr[d] * r * w[d];
}

// ===========================================================================
// WMMA GEMM: C[M,N] = A[M,K]@B[K,N] (+= if ADD). Block tile 32x128, 8 waves,
// each wave owns two 16x16 C tiles (2 wmma / K-step). A staged to LDS as f16;
// B tile DMA'd by TDM (or cooperatively) as f32, stride 129.
// Requires M%32==0, N%128==0, K%32==0.
// ===========================================================================
template <int ADD>
__global__ void k_gemm_t(const float* __restrict__ A, const float* __restrict__ B,
                         float* __restrict__ C, int M, int N, int K) {
    __shared__ _Float16 As[32 * 34];
    __shared__ float    Bsf[32 * 129];
    int t = threadIdx.x;
    int bm = blockIdx.y * 32, bn = blockIdx.x * 128;
    int w = t >> 5, lane = t & 31;
    int m = lane & 15, half = lane >> 4, kb = half * 8;
    int tr = w >> 2, tc = w & 3;
    int arow = t >> 3, ac4 = (t & 7) * 4;

    vfloat8 acc0 = {}, acc1 = {};
    for (int k0 = 0; k0 < K; k0 += 32) {
        float4 av = *(const float4*)(A + (size_t)(bm + arow) * K + k0 + ac4);
        __syncthreads();                      // prior-iteration LDS reads done
#if USE_TDM
        if (w == 0) {
            tdm_load_f32_tile(B + (size_t)k0 * N + bn,
                              (unsigned)(__SIZE_TYPE__)Bsf, N);
        }
#else
        stage_b_coop(B + (size_t)k0 * N + bn, N, t, Bsf);
#endif
        As[arow * 34 + ac4 + 0] = (_Float16)av.x;
        As[arow * 34 + ac4 + 1] = (_Float16)av.y;
        As[arow * 34 + ac4 + 2] = (_Float16)av.z;
        As[arow * 34 + ac4 + 3] = (_Float16)av.w;
#if USE_TDM
        if (w == 0) __builtin_amdgcn_s_wait_tensorcnt(0);
#endif
        __syncthreads();
        vhalf16 a, b0, b1;
        const _Float16* ap = As + (tr * 16 + m) * 34 + kb;
        #pragma unroll
        for (int e = 0; e < 8; e++) a[e] = ap[e];
        #pragma unroll
        for (int e = 0; e < 8; e++) a[8 + e] = ap[16 + e];
        int c0 = tc * 32 + m;
        #pragma unroll
        for (int e = 0; e < 8; e++) {
            b0[e]     = (_Float16)Bsf[(kb + e) * 129 + c0];
            b1[e]     = (_Float16)Bsf[(kb + e) * 129 + c0 + 16];
        }
        #pragma unroll
        for (int e = 0; e < 8; e++) {
            b0[8 + e] = (_Float16)Bsf[(16 + kb + e) * 129 + c0];
            b1[8 + e] = (_Float16)Bsf[(16 + kb + e) * 129 + c0 + 16];
        }
        acc0 = wmma_f16(a, b0, acc0);
        acc1 = wmma_f16(a, b1, acc1);
    }
    float* base = C + (size_t)(bm + tr * 16 + 8 * half) * N + bn + tc * 32 + m;
    #pragma unroll
    for (int r = 0; r < 8; r++) {
        float* p = base + (size_t)r * N;
        if (ADD) { p[0] += acc0[r]; p[16] += acc1[r]; }
        else     { p[0]  = acc0[r]; p[16]  = acc1[r]; }
    }
}

// ===========================================================================
// RoPE (in place)
// ===========================================================================
__global__ void k_rope(float* __restrict__ buf, int nh) {
    int s = blockIdx.x, hh = blockIdx.y, p = threadIdx.x;
    float inv = __powf(10000.f, -(float)(2 * p) / (float)HD);
    float ang = (float)s * inv;
    float c = __cosf(ang), sn = __sinf(ang);
    float* base = buf + ((size_t)s * nh + hh) * HD;
    float x1 = base[2 * p], x2 = base[2 * p + 1];
    base[2 * p]     = x1 * c - x2 * sn;
    base[2 * p + 1] = x1 * sn + x2 * c;
}

// ===========================================================================
// Sliding-window flash attention (one wave per 16-query tile x head).
// ===========================================================================
__global__ void k_attn(const float* __restrict__ q, const float* __restrict__ k,
                       const float* __restrict__ v, float* __restrict__ o) {
    int qb = blockIdx.x, h = blockIdx.y;
    int kvh = h / (NHEAD / NKV);
    int lane = threadIdx.x;
    int q0 = qb * 16;
    int m = lane & 15, half = lane >> 4, kb = half * 8;
    __shared__ _Float16 P[16 * 32];

    vhalf16 qa[2];
    const float* qrow = q + ((size_t)(q0 + m) * NHEAD + h) * HD;
    #pragma unroll
    for (int c = 0; c < 2; c++) {
        #pragma unroll
        for (int e = 0; e < 8; e++) qa[c][e]     = (_Float16)(qrow[c * 32 + kb + e] * 0.125f);
        #pragma unroll
        for (int e = 0; e < 8; e++) qa[c][8 + e] = (_Float16)(qrow[c * 32 + 16 + kb + e] * 0.125f);
    }
    vfloat8 acc0 = {}, acc1 = {}, acc2 = {}, acc3 = {};
    float ms[8], ls[8];
    #pragma unroll
    for (int r = 0; r < 8; r++) { ms[r] = -1e30f; ls[r] = 0.f; }

    int klo = q0 - (WIN - 1); if (klo < 0) klo = 0;
    int kc0 = (klo >> 5) << 5;
    int kc1 = ((q0 + 15) >> 5) << 5;
    for (int kc = kc0; kc <= kc1; kc += 32) {
        vfloat8 s0 = {}, s1 = {};
        #pragma unroll
        for (int tt = 0; tt < 2; tt++) {
            int key = kc + tt * 16 + m;
            const float* krow = k + ((size_t)key * NKV + kvh) * HD;
            #pragma unroll
            for (int c = 0; c < 2; c++) {
                vhalf16 bb;
                #pragma unroll
                for (int e = 0; e < 8; e++) bb[e]     = (_Float16)krow[c * 32 + kb + e];
                #pragma unroll
                for (int e = 0; e < 8; e++) bb[8 + e] = (_Float16)krow[c * 32 + 16 + kb + e];
                if (tt == 0) s0 = wmma_f16(qa[c], bb, s0);
                else         s1 = wmma_f16(qa[c], bb, s1);
            }
        }
        #pragma unroll
        for (int r = 0; r < 8; r++) {
            int i = q0 + r + 8 * half;
            int j0 = kc + m, j1 = kc + 16 + m;
            float a0 = ((j0 <= i) && (i - j0 < WIN)) ? s0[r] : -1e30f;
            float a1 = ((j1 <= i) && (i - j1 < WIN)) ? s1[r] : -1e30f;
            float mx = fmaxf(a0, a1);
            #pragma unroll
            for (int d = 8; d >= 1; d >>= 1) mx = fmaxf(mx, __shfl_xor(mx, d));
            float nm = fmaxf(ms[r], mx);
            float sc = __expf(ms[r] - nm);
            float p0 = (a0 > -1e29f) ? __expf(a0 - nm) : 0.f;
            float p1 = (a1 > -1e29f) ? __expf(a1 - nm) : 0.f;
            float rs = p0 + p1;
            #pragma unroll
            for (int d = 8; d >= 1; d >>= 1) rs += __shfl_xor(rs, d);
            ls[r] = ls[r] * sc + rs;
            ms[r] = nm;
            acc0[r] *= sc; acc1[r] *= sc; acc2[r] *= sc; acc3[r] *= sc;
            int row = r + 8 * half;
            P[row * 32 + m]      = (_Float16)p0;
            P[row * 32 + 16 + m] = (_Float16)p1;
        }
        __syncthreads();
        vhalf16 pa;
        #pragma unroll
        for (int e = 0; e < 8; e++) pa[e]     = P[m * 32 + kb + e];
        #pragma unroll
        for (int e = 0; e < 8; e++) pa[8 + e] = P[m * 32 + 16 + kb + e];
        #pragma unroll
        for (int tt = 0; tt < 4; tt++) {
            vhalf16 vb;
            int hd = tt * 16 + m;
            #pragma unroll
            for (int e = 0; e < 8; e++) {
                int key = kc + kb + e;
                vb[e] = (_Float16)v[((size_t)key * NKV + kvh) * HD + hd];
            }
            #pragma unroll
            for (int e = 0; e < 8; e++) {
                int key = kc + 16 + kb + e;
                vb[8 + e] = (_Float16)v[((size_t)key * NKV + kvh) * HD + hd];
            }
            if (tt == 0) acc0 = wmma_f16(pa, vb, acc0);
            else if (tt == 1) acc1 = wmma_f16(pa, vb, acc1);
            else if (tt == 2) acc2 = wmma_f16(pa, vb, acc2);
            else              acc3 = wmma_f16(pa, vb, acc3);
        }
        __syncthreads();
    }
    #pragma unroll
    for (int r = 0; r < 8; r++) {
        int i = q0 + r + 8 * half;
        float inv = 1.f / ls[r];
        float* orow = o + (size_t)i * (NHEAD * HD) + h * HD;
        orow[0 * 16 + m] = acc0[r] * inv;
        orow[1 * 16 + m] = acc1[r] * inv;
        orow[2 * 16 + m] = acc2[r] * inv;
        orow[3 * 16 + m] = acc3[r] * inv;
    }
}

// ===========================================================================
// Router gate: top-2 + softmax weights (one wave per token)
// ===========================================================================
__global__ void k_gate(const float* __restrict__ h, const float* __restrict__ gw,
                       int* __restrict__ topi, float* __restrict__ wts) {
    int s = blockIdx.x, lane = threadIdx.x;
    float acc[NEXP] = {};
    const float* hr = h + (size_t)s * DIM;
    for (int d = lane; d < DIM; d += 32) {
        float hv = hr[d];
        const float* g = gw + (size_t)d * NEXP;
        #pragma unroll
        for (int e = 0; e < NEXP; e++) acc[e] += hv * g[e];
    }
    #pragma unroll
    for (int e = 0; e < NEXP; e++) {
        #pragma unroll
        for (int mk = 16; mk >= 1; mk >>= 1) acc[e] += __shfl_xor(acc[e], mk);
    }
    if (lane == 0) {
        int i0 = 0;
        for (int e = 1; e < NEXP; e++) if (acc[e] > acc[i0]) i0 = e;
        int i1 = (i0 == 0) ? 1 : 0;
        for (int e = 0; e < NEXP; e++) if (e != i0 && acc[e] > acc[i1]) i1 = e;
        float ex = __expf(acc[i1] - acc[i0]);
        float den = 1.f + ex;
        topi[s * 2] = i0;        topi[s * 2 + 1] = i1;
        wts[s * 2]  = 1.f / den; wts[s * 2 + 1] = ex / den;
    }
}

__global__ void k_route(const int* __restrict__ topi, int* __restrict__ cnt,
                        int* __restrict__ list) {
    int idx = blockIdx.x * blockDim.x + threadIdx.x;
    if (idx >= 2 * SEQ) return;
    int e = topi[idx];
    int pos = atomicAdd(&cnt[e], 1);
    list[e * SEQ + pos] = idx;
}

// ===========================================================================
// Grouped MoE up-projection: act[asg,:] = silu(h@w1[e]) * (h@w3[e])
// grid = (FFD/128, SEQ/16, NEXP); TDM-staged weight tiles.
// ===========================================================================
__global__ void k_moe_up(const float* __restrict__ hb, const float* __restrict__ w1l,
                         const float* __restrict__ w3l, const int* __restrict__ cnt,
                         const int* __restrict__ list, float* __restrict__ act) {
    __shared__ int   asg[16];
    __shared__ float B1s[32 * 129];
    __shared__ float B3s[32 * 129];
    int e = blockIdx.z, tile = blockIdx.y;
    int ce = cnt[e];
    if (tile * 16 >= ce) return;
    int t = threadIdx.x;
    if (t < 16) {
        int idx = tile * 16 + t;
        asg[t] = (idx < ce) ? list[e * SEQ + idx] : -1;
    }
    __syncthreads();
    int w = t >> 5, lane = t & 31;
    int m = lane & 15, half = lane >> 4, kb = half * 8;
    int bn = blockIdx.x * 128;
    const float* B1 = w1l + (size_t)e * DIM * FFD + bn;
    const float* B3 = w3l + (size_t)e * DIM * FFD + bn;
    int a_m = asg[m];
    const float* arow = hb + (size_t)((a_m < 0) ? 0 : (a_m >> 1)) * DIM;
    int col = w * 16 + m;
    vfloat8 u = {}, g = {};
    for (int k0 = 0; k0 < DIM; k0 += 32) {
        __syncthreads();
#if USE_TDM
        if (w == 0) {
            tdm_load_f32_tile(B1 + (size_t)k0 * FFD, (unsigned)(__SIZE_TYPE__)B1s, FFD);
            tdm_load_f32_tile(B3 + (size_t)k0 * FFD, (unsigned)(__SIZE_TYPE__)B3s, FFD);
            __builtin_amdgcn_s_wait_tensorcnt(0);
        }
#else
        stage_b_coop(B1 + (size_t)k0 * FFD, FFD, t, B1s);
        stage_b_coop(B3 + (size_t)k0 * FFD, FFD, t, B3s);
#endif
        __syncthreads();
        vhalf16 a, b1, b3;
        #pragma unroll
        for (int e2 = 0; e2 < 8; e2++) a[e2]     = (_Float16)arow[k0 + kb + e2];
        #pragma unroll
        for (int e2 = 0; e2 < 8; e2++) a[8 + e2] = (_Float16)arow[k0 + 16 + kb + e2];
        #pragma unroll
        for (int e2 = 0; e2 < 8; e2++) {
            b1[e2]     = (_Float16)B1s[(kb + e2) * 129 + col];
            b3[e2]     = (_Float16)B3s[(kb + e2) * 129 + col];
        }
        #pragma unroll
        for (int e2 = 0; e2 < 8; e2++) {
            b1[8 + e2] = (_Float16)B1s[(16 + kb + e2) * 129 + col];
            b3[8 + e2] = (_Float16)B3s[(16 + kb + e2) * 129 + col];
        }
        u = wmma_f16(a, b1, u);
        g = wmma_f16(a, b3, g);
    }
    #pragma unroll
    for (int r = 0; r < 8; r++) {
        int rl = r + 8 * half;
        int a_r = asg[rl];
        if (a_r >= 0) {
            float uu = u[r];
            act[(size_t)a_r * FFD + bn + col] = (uu / (1.f + __expf(-uu))) * g[r];
        }
    }
}

// ===========================================================================
// Grouped MoE down-projection + weighted scatter-add into residual stream.
// grid = (DIM/128, SEQ/16, NEXP)
// ===========================================================================
__global__ void k_moe_down(const float* __restrict__ act, const float* __restrict__ w2l,
                           const float* __restrict__ wts, const int* __restrict__ cnt,
                           const int* __restrict__ list, float* __restrict__ x) {
    __shared__ int   asg[16];
    __shared__ float Bsf[32 * 129];
    int e = blockIdx.z, tile = blockIdx.y;
    int ce = cnt[e];
    if (tile * 16 >= ce) return;
    int t = threadIdx.x;
    if (t < 16) {
        int idx = tile * 16 + t;
        asg[t] = (idx < ce) ? list[e * SEQ + idx] : -1;
    }
    __syncthreads();
    int w = t >> 5, lane = t & 31;
    int m = lane & 15, half = lane >> 4, kb = half * 8;
    int bn = blockIdx.x * 128;
    const float* B = w2l + (size_t)e * FFD * DIM + bn;
    int a_m = asg[m];
    const float* arow = act + (size_t)((a_m < 0) ? 0 : a_m) * FFD;
    int col = w * 16 + m;
    vfloat8 acc = {};
    for (int k0 = 0; k0 < FFD; k0 += 32) {
        __syncthreads();
#if USE_TDM
        if (w == 0) {
            tdm_load_f32_tile(B + (size_t)k0 * DIM, (unsigned)(__SIZE_TYPE__)Bsf, DIM);
            __builtin_amdgcn_s_wait_tensorcnt(0);
        }
#else
        stage_b_coop(B + (size_t)k0 * DIM, DIM, t, Bsf);
#endif
        __syncthreads();
        vhalf16 a, b;
        #pragma unroll
        for (int e2 = 0; e2 < 8; e2++) a[e2]     = (_Float16)arow[k0 + kb + e2];
        #pragma unroll
        for (int e2 = 0; e2 < 8; e2++) a[8 + e2] = (_Float16)arow[k0 + 16 + kb + e2];
        #pragma unroll
        for (int e2 = 0; e2 < 8; e2++) b[e2]     = (_Float16)Bsf[(kb + e2) * 129 + col];
        #pragma unroll
        for (int e2 = 0; e2 < 8; e2++) b[8 + e2] = (_Float16)Bsf[(16 + kb + e2) * 129 + col];
        acc = wmma_f16(a, b, acc);
    }
    #pragma unroll
    for (int r = 0; r < 8; r++) {
        int rl = r + 8 * half;
        int a_r = asg[rl];
        if (a_r >= 0) {
            float wt = wts[a_r];
            atomicAdd(&x[(size_t)(a_r >> 1) * DIM + bn + col], acc[r] * wt);
        }
    }
}

// ===========================================================================
// Host-side orchestration
// ===========================================================================
extern "C" void kernel_launch(void* const* d_in, const int* in_sizes, int n_in,
                              void* d_out, int out_size, void* d_ws, size_t ws_size,
                              hipStream_t stream) {
    const int*   tokens       = (const int*)d_in[0];
    const float* tok_emb      = (const float*)d_in[1];
    const float* out_w        = (const float*)d_in[2];
    const float* final_norm_w = (const float*)d_in[3];
    const float* attn_norm_w  = (const float*)d_in[4];
    const float* ffn_norm_w   = (const float*)d_in[5];
    const float* wq           = (const float*)d_in[6];
    const float* wk           = (const float*)d_in[7];
    const float* wv           = (const float*)d_in[8];
    const float* wo           = (const float*)d_in[9];
    const float* gate_w       = (const float*)d_in[10];
    const float* w1           = (const float*)d_in[11];
    const float* w2           = (const float*)d_in[12];
    const float* w3           = (const float*)d_in[13];
    float* out = (float*)d_out;

    // Workspace layout (~48.3 MB)
    float* ws = (float*)d_ws;
    float* x    = ws;                                  // SEQ*DIM
    float* hbuf = x    + (size_t)SEQ * DIM;            // SEQ*DIM
    float* qbuf = hbuf + (size_t)SEQ * DIM;            // SEQ*NHEAD*HD
    float* kbuf = qbuf + (size_t)SEQ * NHEAD * HD;     // SEQ*NKV*HD
    float* vbuf = kbuf + (size_t)SEQ * NKV * HD;       // SEQ*NKV*HD
    float* obuf = vbuf + (size_t)SEQ * NKV * HD;       // SEQ*NHEAD*HD
    float* act  = obuf + (size_t)SEQ * NHEAD * HD;     // 2*SEQ*FFD
    float* wts  = act  + (size_t)2 * SEQ * FFD;        // 2*SEQ
    int* topi = (int*)(wts + 2 * SEQ);                 // 2*SEQ
    int* cnt  = topi + 2 * SEQ;                        // NEXP
    int* list = cnt + NEXP;                            // NEXP*SEQ

    k_embed<<<SEQ, 256, 0, stream>>>(tokens, tok_emb, x);

    for (int l = 0; l < NLAYR; l++) {
        // ---- attention block ----
        k_rmsnorm<<<SEQ, 256, 0, stream>>>(x, attn_norm_w + (size_t)l * DIM, hbuf);
        k_gemm_t<0><<<dim3(DIM / 128, SEQ / 32), 256, 0, stream>>>(
            hbuf, wq + (size_t)l * DIM * NHEAD * HD, qbuf, SEQ, NHEAD * HD, DIM);
        k_gemm_t<0><<<dim3((NKV * HD) / 128, SEQ / 32), 256, 0, stream>>>(
            hbuf, wk + (size_t)l * DIM * NKV * HD, kbuf, SEQ, NKV * HD, DIM);
        k_gemm_t<0><<<dim3((NKV * HD) / 128, SEQ / 32), 256, 0, stream>>>(
            hbuf, wv + (size_t)l * DIM * NKV * HD, vbuf, SEQ, NKV * HD, DIM);
        k_rope<<<dim3(SEQ, NHEAD), 32, 0, stream>>>(qbuf, NHEAD);
        k_rope<<<dim3(SEQ, NKV), 32, 0, stream>>>(kbuf, NKV);
        k_attn<<<dim3(SEQ / 16, NHEAD), 32, 0, stream>>>(qbuf, kbuf, vbuf, obuf);
        k_gemm_t<1><<<dim3(DIM / 128, SEQ / 32), 256, 0, stream>>>(
            obuf, wo + (size_t)l * NHEAD * HD * DIM, x, SEQ, DIM, NHEAD * HD);

        // ---- MoE block ----
        k_rmsnorm<<<SEQ, 256, 0, stream>>>(x, ffn_norm_w + (size_t)l * DIM, hbuf);
        k_gate<<<SEQ, 32, 0, stream>>>(hbuf, gate_w + (size_t)l * DIM * NEXP, topi, wts);
        (void)hipMemsetAsync(cnt, 0, NEXP * sizeof(int), stream);
        k_route<<<(2 * SEQ + 255) / 256, 256, 0, stream>>>(topi, cnt, list);
        k_moe_up<<<dim3(FFD / 128, SEQ / 16, NEXP), 256, 0, stream>>>(
            hbuf, w1 + (size_t)l * NEXP * DIM * FFD,
            w3 + (size_t)l * NEXP * DIM * FFD, cnt, list, act);
        k_moe_down<<<dim3(DIM / 128, SEQ / 16, NEXP), 256, 0, stream>>>(
            act, w2 + (size_t)l * NEXP * FFD * DIM, wts, cnt, list, x);
    }

    // ---- final norm + logits ----
    k_rmsnorm<<<SEQ, 256, 0, stream>>>(x, final_norm_w, hbuf);
    k_gemm_t<0><<<dim3(VOCAB / 128, SEQ / 32), 256, 0, stream>>>(
        hbuf, out_w, out, SEQ, VOCAB, DIM);
}
